// Model_446676599348
// MI455X (gfx1250) — matrix operand order repaired and checked by reading, not compiled, not run on hardware
//
#include <hip/hip_runtime.h>
#include <hip/hip_bf16.h>

// ---------------- types ----------------
typedef __attribute__((ext_vector_type(16))) __bf16        v16bf;
typedef __attribute__((ext_vector_type(8)))  float         v8f;
typedef __attribute__((ext_vector_type(8)))  unsigned int  v8u;
typedef __attribute__((ext_vector_type(4)))  float         fvec4;

__device__ __forceinline__ unsigned short f2bf(float f) {
    union { float f; unsigned int u; } c; c.f = f;
    unsigned int r = c.u + 0x7FFFu + ((c.u >> 16) & 1u);   // round-to-nearest-even
    return (unsigned short)(r >> 16);
}

// ============================================================
// Kernel 1: per-(b,j) argmax over 256x256 heatmap -> normalized grid
// Bandwidth-dominant stage: 256 MB read-once stream -> non-temporal loads.
// ============================================================
__launch_bounds__(256)
__global__ void argmax_kernel(const float* __restrict__ hm, float* __restrict__ grid) {
    const int bj  = blockIdx.x;
    const int tid = threadIdx.x;
    const fvec4* p = (const fvec4*)(hm + (size_t)bj * 65536);

    float bv = -INFINITY; int bi = 0;
#pragma unroll 4
    for (int i = 0; i < 64; ++i) {
        const int q = tid + i * 256;          // coalesced 16B stream
        const fvec4 v = __builtin_nontemporal_load(p + q);
        const int base = q * 4;
        if (v.x > bv) { bv = v.x; bi = base;     }
        if (v.y > bv) { bv = v.y; bi = base + 1; }
        if (v.z > bv) { bv = v.z; bi = base + 2; }
        if (v.w > bv) { bv = v.w; bi = base + 3; }
    }

    __shared__ float sv[256];
    __shared__ int   si[256];
    sv[tid] = bv; si[tid] = bi;
    __syncthreads();
    for (int s = 128; s > 0; s >>= 1) {
        if (tid < s) {
            const float ov = sv[tid + s]; const int oi = si[tid + s];
            if (ov > sv[tid] || (ov == sv[tid] && oi < si[tid])) { sv[tid] = ov; si[tid] = oi; }
        }
        __syncthreads();
    }
    if (tid == 0) {
        const int idx = si[0];
        const float x = (float)(idx & 255);
        const float y = (float)(idx >> 8);
        grid[bj * 2 + 0] = (x - 127.5f) / 127.5f;
        grid[bj * 2 + 1] = (y - 127.5f) / 127.5f;
    }
}

// ============================================================
// Kernel 2: gather samples. Nearest pos_embed -> out ch 224..255.
// Bilinear feat1/2/3 -> bf16 feature matrix X[1024][1024] (K-padded).
// ============================================================
__device__ __forceinline__ void bilin_setup(float g, int W, int& i0, int& i1, float& w) {
    const float s = (g + 1.0f) * 0.5f * (float)(W - 1);
    const float f = floorf(s);
    w  = s - f;
    int fi = (int)f;
    i0 = min(max(fi,     0), W - 1);
    i1 = min(max(fi + 1, 0), W - 1);
}

__launch_bounds__(256)
__global__ void sample_kernel(const float* __restrict__ grid,
                              const float* __restrict__ pos_embed,
                              const float* __restrict__ f1,
                              const float* __restrict__ f2,
                              const float* __restrict__ f3,
                              unsigned short* __restrict__ X,   // stride 1024, cols 992..1023 zero
                              float* __restrict__ out) {
    const int bj  = blockIdx.x;       // b*64 + j
    const int b   = bj >> 6;
    const int j   = bj & 63;
    const int tid = threadIdx.x;
    const float gx = grid[bj * 2 + 0];
    const float gy = grid[bj * 2 + 1];

    // positions: nearest sample on 256x256 (round-half-even like jnp.round)
    if (tid < 32) {
        const float xs = (gx + 1.0f) * 0.5f * 255.0f;
        const float ys = (gy + 1.0f) * 0.5f * 255.0f;
        const int ix = min(max((int)rintf(xs), 0), 255);
        const int iy = min(max((int)rintf(ys), 0), 255);
        out[((b * 256) + 224 + tid) * 64 + j] =
            pos_embed[(((size_t)b * 32 + tid) * 256 + iy) * 256 + ix];
        // zero the K-padding columns (992..1023)
        X[(size_t)bj * 1024 + 992 + tid] = 0;
    }

    int x0, x1, y0, y1; float wx, wy;

    // feat1: C=96, 128x128
    bilin_setup(gx, 128, x0, x1, wx);
    bilin_setup(gy, 128, y0, y1, wy);
    {
        const float* fb = f1 + (size_t)b * 96 * 128 * 128;
        for (int c = tid; c < 96; c += 256) {
            const float* fc = fb + (size_t)c * 128 * 128;
            const float v = fc[y0 * 128 + x0] * (1.f - wx) * (1.f - wy)
                          + fc[y0 * 128 + x1] * wx        * (1.f - wy)
                          + fc[y1 * 128 + x0] * (1.f - wx) * wy
                          + fc[y1 * 128 + x1] * wx        * wy;
            X[(size_t)bj * 1024 + c] = f2bf(v);
        }
    }
    // feat2: C=384, 64x64
    bilin_setup(gx, 64, x0, x1, wx);
    bilin_setup(gy, 64, y0, y1, wy);
    {
        const float* fb = f2 + (size_t)b * 384 * 64 * 64;
        for (int c = tid; c < 384; c += 256) {
            const float* fc = fb + (size_t)c * 64 * 64;
            const float v = fc[y0 * 64 + x0] * (1.f - wx) * (1.f - wy)
                          + fc[y0 * 64 + x1] * wx        * (1.f - wy)
                          + fc[y1 * 64 + x0] * (1.f - wx) * wy
                          + fc[y1 * 64 + x1] * wx        * wy;
            X[(size_t)bj * 1024 + 96 + c] = f2bf(v);
        }
    }
    // feat3: C=512, 32x32
    bilin_setup(gx, 32, x0, x1, wx);
    bilin_setup(gy, 32, y0, y1, wy);
    {
        const float* fb = f3 + (size_t)b * 512 * 32 * 32;
        for (int c = tid; c < 512; c += 256) {
            const float* fc = fb + (size_t)c * 32 * 32;
            const float v = fc[y0 * 32 + x0] * (1.f - wx) * (1.f - wy)
                          + fc[y0 * 32 + x1] * wx        * (1.f - wy)
                          + fc[y1 * 32 + x0] * (1.f - wx) * wy
                          + fc[y1 * 32 + x1] * wx        * wy;
            X[(size_t)bj * 1024 + 480 + c] = f2bf(v);
        }
    }
}

// ============================================================
// Kernel 3: f32 (K,N) -> bf16 (Np,Kp) transposed + zero-padded weights.
// Padding rows/cols are zero => no effect on dot products.
// ============================================================
__launch_bounds__(256)
__global__ void cvt_t_kernel(const float* __restrict__ W, unsigned short* __restrict__ Wt,
                             int K, int N, int Kp, int Np) {
    const int i = blockIdx.x * 256 + threadIdx.x;
    if (i < Kp * Np) {
        const int n = i / Kp;
        const int k = i - n * Kp;
        Wt[i] = (k < K && n < N) ? f2bf(W[(size_t)k * N + n]) : (unsigned short)0;
    }
}

// ============================================================
// Kernel 4: bf16 WMMA GEMM:  Y = relu(A[1024,Kp] @ Bt[Np,Kp]^T + bias)
// Block: 256 threads = 8 waves. Block tile M=128, N=64, K-tile=64.
// Software-pipelined: prefetch next k-tile into registers (global_load_b128
// clause) while computing current tile from LDS. Per LDS stage: 2 WMMA
// k-steps x 4 n-tiles = 8 WMMAs; fragments all loaded before the WMMA burst
// so only one s_wait_dscnt gates four back-to-back v_wmma ops.
// Fragment layouts used here (A and B share one k order):
//   A 16-bit 16x32: row = lane&15; lanes 0-15 K=0..15, lanes 16-31 K=16..31
//   B 16-bit 32x16: lane = column; lanes 0-15 K=0..15, lanes 16-31 K=16..31 (code's choice, not ISA order)
//   C/D f32 16x16:  vgpr r: lanes 0-15 -> M=r, lanes 16-31 -> M=8+r; N=lane&15
// ============================================================
template <bool FINAL>
__launch_bounds__(256)
__global__ void gemm_bf16_kernel(const unsigned short* __restrict__ A,
                                 const unsigned short* __restrict__ Bt,   // (Np,Kp) bf16
                                 const float* __restrict__ bias,
                                 unsigned short* __restrict__ Ybf,        // stride Nl
                                 float* __restrict__ out,
                                 int Kp, int Nl) {
    __shared__ uint4 ldsA4[1024];   // 128 x 64 bf16, row-major  (16 KB)
    __shared__ uint4 ldsB4[512];    //  64 x 64 bf16, n-major    ( 8 KB)

    const int tid    = threadIdx.x;
    const int lane   = tid & 31;
    const int w      = tid >> 5;
    const int mBlock = blockIdx.y * 128;
    const int nBlock = blockIdx.x * 64;

    // ---- per-thread staging addresses (16B chunks; Kp*2 bytes is 128B-multiple) ----
    const int rA = tid >> 3;               // rows 0..31 (+32 per chunk)
    const int cA = (tid & 7) * 8;          // bf16 column within 64-wide k-tile
    const unsigned short* aP0 = A  + (size_t)(mBlock + rA)      * Kp + cA;
    const unsigned short* aP1 = A  + (size_t)(mBlock + rA + 32) * Kp + cA;
    const unsigned short* aP2 = A  + (size_t)(mBlock + rA + 64) * Kp + cA;
    const unsigned short* aP3 = A  + (size_t)(mBlock + rA + 96) * Kp + cA;
    const unsigned short* bP0 = Bt + (size_t)(nBlock + rA)      * Kp + cA;
    const unsigned short* bP1 = Bt + (size_t)(nBlock + rA + 32) * Kp + cA;

    // ---- prefetch k-tile 0 into registers ----
    uint4 ra0 = *(const uint4*)aP0;
    uint4 ra1 = *(const uint4*)aP1;
    uint4 ra2 = *(const uint4*)aP2;
    uint4 ra3 = *(const uint4*)aP3;
    uint4 rb0 = *(const uint4*)bP0;
    uint4 rb1 = *(const uint4*)bP1;

    v8f acc[4];
#pragma unroll
    for (int t = 0; t < 4; ++t)
#pragma unroll
        for (int e = 0; e < 8; ++e) acc[t][e] = 0.0f;

    unsigned int* ldsAu = (unsigned int*)ldsA4;
    unsigned int* ldsBu = (unsigned int*)ldsB4;
    const int kh    = lane >> 4;
    const int nLane = lane & 15;
    const int arow  = (w * 16 + (lane & 15)) * 32;   // dwords per LDS-A row = 32

    for (int kk = 0; kk < Kp; kk += 64) {
        __syncthreads();
        ldsA4[tid]       = ra0;
        ldsA4[tid + 256] = ra1;
        ldsA4[tid + 512] = ra2;
        ldsA4[tid + 768] = ra3;
        ldsB4[tid]       = rb0;
        ldsB4[tid + 256] = rb1;
        __syncthreads();

        // prefetch next k-tile (clause of global_load_b128; overlaps WMMA below)
        if (kk + 64 < Kp) {
            ra0 = *(const uint4*)(aP0 + kk + 64);
            ra1 = *(const uint4*)(aP1 + kk + 64);
            ra2 = *(const uint4*)(aP2 + kk + 64);
            ra3 = *(const uint4*)(aP3 + kk + 64);
            rb0 = *(const uint4*)(bP0 + kk + 64);
            rb1 = *(const uint4*)(bP1 + kk + 64);
        }

        // ---- two WMMA k-steps per LDS stage ----
#pragma unroll
        for (int ks = 0; ks < 2; ++ks) {
            const int kbase = ks * 16;               // dword offset of this 32-k step
            // A fragment (contiguous K halves: lanes 0-15 K=0..15, lanes 16-31 K=16..31, matching B)
            v8u au;
#pragma unroll
            for (int i = 0; i < 4; ++i) {
                au[i]     = ldsAu[arow + kbase + kh * 8 + i];
                au[4 + i] = ldsAu[arow + kbase + kh * 8 + 4 + i];
            }
            // all four B fragments loaded before the WMMA burst
            v8u bu0, bu1, bu2, bu3;
            const int brow = nLane * 32 + kbase + kh * 8;
#pragma unroll
            for (int i = 0; i < 8; ++i) bu0[i] = ldsBu[brow          + i];
#pragma unroll
            for (int i = 0; i < 8; ++i) bu1[i] = ldsBu[brow + 16 * 32 + i];
#pragma unroll
            for (int i = 0; i < 8; ++i) bu2[i] = ldsBu[brow + 32 * 32 + i];
#pragma unroll
            for (int i = 0; i < 8; ++i) bu3[i] = ldsBu[brow + 48 * 32 + i];

            const v16bf a  = __builtin_bit_cast(v16bf, au);
            acc[0] = __builtin_amdgcn_wmma_f32_16x16x32_bf16(
                false, a, false, __builtin_bit_cast(v16bf, bu0), (short)0, acc[0], false, false);
            acc[1] = __builtin_amdgcn_wmma_f32_16x16x32_bf16(
                false, a, false, __builtin_bit_cast(v16bf, bu1), (short)0, acc[1], false, false);
            acc[2] = __builtin_amdgcn_wmma_f32_16x16x32_bf16(
                false, a, false, __builtin_bit_cast(v16bf, bu2), (short)0, acc[2], false, false);
            acc[3] = __builtin_amdgcn_wmma_f32_16x16x32_bf16(
                false, a, false, __builtin_bit_cast(v16bf, bu3), (short)0, acc[3], false, false);
        }
    }

    // ---- epilogue: bias + relu, store ----
    const int mOut     = mBlock + w * 16 + (lane >> 4) * 8;
    const int nOutBase = nBlock + (lane & 15);
#pragma unroll
    for (int nt = 0; nt < 4; ++nt) {
        if (nBlock + nt * 16 < Nl) {                  // uniform per block (Nl % 16 == 0)
            const int col = nOutBase + nt * 16;
            const float bv = bias[col];
#pragma unroll
            for (int r = 0; r < 8; ++r) {
                float v = fmaxf(acc[nt][r] + bv, 0.0f);
                const int row = mOut + r;
                if (FINAL) {
                    // row = b*64 + j ; out[(b*256 + col)*64 + j]
                    out[(((row >> 6) * 256) + col) * 64 + (row & 63)] = v;
                } else {
                    Ybf[(size_t)row * Nl + col] = f2bf(v);
                }
            }
        }
    }
}

// ============================================================
// Launch
// ============================================================
extern "C" void kernel_launch(void* const* d_in, const int* in_sizes, int n_in,
                              void* d_out, int out_size, void* d_ws, size_t ws_size,
                              hipStream_t stream) {
    const float* heatmap   = (const float*)d_in[0];
    const float* pos_embed = (const float*)d_in[1];
    const float* feat1     = (const float*)d_in[2];
    const float* feat2     = (const float*)d_in[3];
    const float* feat3     = (const float*)d_in[4];
    const float* W1 = (const float*)d_in[5];  const float* b1 = (const float*)d_in[6];
    const float* W2 = (const float*)d_in[7];  const float* b2 = (const float*)d_in[8];
    const float* W3 = (const float*)d_in[9];  const float* b3 = (const float*)d_in[10];
    const float* W4 = (const float*)d_in[11]; const float* b4 = (const float*)d_in[12];
    float* out = (float*)d_out;

    // workspace carve-up (~10 MB total)
    char* ws = (char*)d_ws;
    size_t off = 0;
    auto carve = [&](size_t bytes) -> void* {
        void* p = ws + off;
        off = (off + bytes + 255) & ~(size_t)255;
        return p;
    };
    float*          grid = (float*)         carve((size_t)1024 * 2 * 4);
    unsigned short* X    = (unsigned short*)carve((size_t)1024 * 1024 * 2);  // K-padded 992->1024
    unsigned short* W1t  = (unsigned short*)carve((size_t)1024 * 1024 * 2);  // (N=1024, Kp=1024)
    unsigned short* W2t  = (unsigned short*)carve((size_t)512 * 1024 * 2);   // (512, 1024)
    unsigned short* W3t  = (unsigned short*)carve((size_t)256 * 512 * 2);    // (256, 512)
    unsigned short* W4t  = (unsigned short*)carve((size_t)256 * 256 * 2);    // (Np=256, 256), rows 224..255 zero
    unsigned short* Y1   = (unsigned short*)carve((size_t)1024 * 1024 * 2);
    unsigned short* Y2   = (unsigned short*)carve((size_t)1024 * 512 * 2);
    unsigned short* Y3   = (unsigned short*)carve((size_t)1024 * 256 * 2);

    // 1) argmax over heatmap (bandwidth-dominant: 256 MB NT stream)
    argmax_kernel<<<1024, 256, 0, stream>>>(heatmap, grid);

    // 2) gather: positions -> out[ch 224..255], bilinear feats -> X (bf16, K-padded)
    sample_kernel<<<1024, 256, 0, stream>>>(grid, pos_embed, feat1, feat2, feat3, X, out);

    // 3) weight conversion + transpose + pad: f32 (K,N) -> bf16 (Np,Kp)
    cvt_t_kernel<<<(1024 * 1024 + 255) / 256, 256, 0, stream>>>(W1, W1t, 992, 1024, 1024, 1024);
    cvt_t_kernel<<<(512 * 1024 + 255) / 256, 256, 0, stream>>>(W2, W2t, 1024, 512, 1024, 512);
    cvt_t_kernel<<<(256 * 512 + 255) / 256, 256, 0, stream>>>(W3, W3t, 512, 256, 512, 256);
    cvt_t_kernel<<<(256 * 256 + 255) / 256, 256, 0, stream>>>(W4, W4t, 256, 224, 256, 256);

    // 4) WMMA MLP chain (M=1024; block tile 128x64, k-tile 64)
    gemm_bf16_kernel<false><<<dim3(1024 / 64, 8), 256, 0, stream>>>(
        X,  W1t, b1, Y1, nullptr, 1024, 1024);
    gemm_bf16_kernel<false><<<dim3(512 / 64, 8), 256, 0, stream>>>(
        Y1, W2t, b2, Y2, nullptr, 1024, 512);
    gemm_bf16_kernel<false><<<dim3(256 / 64, 8), 256, 0, stream>>>(
        Y2, W3t, b3, Y3, nullptr, 512, 256);
    gemm_bf16_kernel<true><<<dim3(256 / 64, 8), 256, 0, stream>>>(
        Y3, W4t, b4, nullptr, out, 256, 224);
}
